// GnBlock_14147622273442
// MI455X (gfx1250) — matrix operand order repaired and checked by reading, not compiled, not run on hardware
//
#include <hip/hip_runtime.h>
#include <hip/hip_bf16.h>
#include <cstdint>
#include <math.h>

// Problem constants (fixed by the reference)
#define Nn 50000
#define Ee 800000
#define Dd 128

typedef __attribute__((ext_vector_type(16))) __bf16 v16bf;
typedef __attribute__((ext_vector_type(8)))  __bf16 v8bf;
typedef __attribute__((ext_vector_type(8)))  float  v8f;

// ---------------- LDS layout (byte offsets into dynamic shared memory) ----
// All three layer weights stay resident for the whole block (CDNA5: 320KB/WGP).
#define OFF_W1   0              // bf16 up to 384x128   (98304 B)
#define OFF_W2   98304          // bf16 128x128         (32768 B)
#define OFF_W3   131072         // bf16 128x128         (32768 B)
#define OFF_A    163840         // bf16 up to 128x384   (98304 B)
#define OFF_H    262144         // bf16 128x128         (32768 B)
#define OFF_B1   294912         // f32[128]
#define OFF_B2   295424         // f32[128]
#define OFF_B3   295936         // f32[128]
#define OFF_GAM  296448         // f32[128]
#define OFF_BET  296960         // f32[128]
#define OFF_SIDX 297472         // i32[128]
#define OFF_RIDX 297984         // i32[128]
#define OFF_CSUM 298496         // f32[128] (node kernel)
#define OFF_CSQ  299008         // f32[128] (node kernel)
#define OFF_PS   299520         // f32[128][2] LN row partial sums
#define OFF_PQ   300544         // f32[128][2] LN row partial sumsq
#define SMEM_BYTES 301568       // < 320KB WGP LDS

__device__ __forceinline__ v16bf cat8(v8bf lo, v8bf hi) {
  return __builtin_shufflevector(lo, hi, 0,1,2,3,4,5,6,7,8,9,10,11,12,13,14,15);
}

__device__ __forceinline__ void zero_acc2(v8f acc[2][4]) {
#pragma unroll
  for (int rt = 0; rt < 2; ++rt)
#pragma unroll
    for (int ct = 0; ct < 4; ++ct)
#pragma unroll
      for (int g = 0; g < 8; ++g) acc[rt][ct][g] = 0.0f;
}

// One wave computes a 32-row x 64-col tile: 2 row-tiles x 4 col-tiles.
// Each B fragment is loaded once and feeds TWO WMMAs (both row tiles), so the
// LDS-load : WMMA ratio drops from 2.25:1 to 1.5:1 and each s_wait_dscnt
// covers two matrix ops.
// A-fragment layout per CDNA5 ISA (16-bit A 16x32): lanes 0-15 hold K {0..7,16..23},
// lanes 16-31 hold K {8..15,24..31} of row M = lane&15.
__device__ __forceinline__ void gemm_rows32(const __bf16* __restrict__ A, int lda,
                                            const __bf16* __restrict__ W, int colb,
                                            int kChunks, int lane, v8f acc[2][4]) {
  const int l = lane & 15;
  const int half = lane >> 4;
  for (int c = 0; c < kChunks; ++c) {
    const __bf16* ap0 = A + (size_t)l * lda + c * 32 + half * 8;
    const __bf16* ap1 = ap0 + (size_t)16 * lda;
    v16bf af0 = cat8(*(const v8bf*)ap0, *(const v8bf*)(ap0 + 16));
    v16bf af1 = cat8(*(const v8bf*)ap1, *(const v8bf*)(ap1 + 16));
    const __bf16* wr = W + (size_t)(c * 32 + half * 8) * Dd + colb + l;  // lane&15 = N col, K in table order
#pragma unroll
    for (int ct = 0; ct < 4; ++ct) {
      const __bf16* bp = wr + ct * 16;
      v16bf bf;
#pragma unroll
      for (int s = 0; s < 8; ++s) { bf[s] = bp[(size_t)s * Dd]; bf[8 + s] = bp[(size_t)(16 + s) * Dd]; }
      acc[0][ct] = __builtin_amdgcn_wmma_f32_16x16x32_bf16(
          false, af0, false, bf, (short)0, acc[0][ct], false, false);
      acc[1][ct] = __builtin_amdgcn_wmma_f32_16x16x32_bf16(
          false, af1, false, bf, (short)0, acc[1][ct], false, false);
    }
  }
}

// bias + SiLU, store bf16 tile to LDS (C layout: VGPR g -> row g (+8 hi lanes))
__device__ __forceinline__ void silu_store2(const v8f acc[2][4], const float* __restrict__ bias,
                                            __bf16* __restrict__ out, int ldo,
                                            int rbase, int colb, int lane) {
  const int l = lane & 15;
  const int mbase = (lane >> 4) * 8;
#pragma unroll
  for (int rt = 0; rt < 2; ++rt)
#pragma unroll
    for (int ct = 0; ct < 4; ++ct) {
      const int col = colb + ct * 16 + l;
      const float bia = bias[col];
#pragma unroll
      for (int g = 0; g < 8; ++g) {
        float v = acc[rt][ct][g] + bia;
        v = v / (1.0f + __expf(-v));               // SiLU
        out[(size_t)(rbase + rt * 16 + mbase + g) * ldo + col] = (__bf16)v;
      }
    }
}

// ---------------------------------------------------------------- prep ----
__global__ void prep_weights(const float* __restrict__ ew1, const float* __restrict__ ew2,
                             const float* __restrict__ ew3, const float* __restrict__ nw1,
                             const float* __restrict__ nw2, const float* __restrict__ nw3,
                             __bf16* __restrict__ wbf) {
  const int S1 = 384 * 128, S2 = S1 + 128 * 128, S3 = S2 + 128 * 128;
  const int S4 = S3 + 256 * 128, S5 = S4 + 128 * 128, S6 = S5 + 128 * 128;
  int i = blockIdx.x * blockDim.x + threadIdx.x;
  if (i >= S6) return;
  float v;
  if (i < S1)      v = ew1[i];
  else if (i < S2) v = ew2[i - S1];
  else if (i < S3) v = ew3[i - S2];
  else if (i < S4) v = nw1[i - S3];
  else if (i < S5) v = nw2[i - S4];
  else             v = nw3[i - S5];
  wbf[i] = (__bf16)v;
}

__global__ void zero_ws(float* __restrict__ agg, float* __restrict__ stats) {
  size_t i = (size_t)blockIdx.x * blockDim.x + threadIdx.x;
  if (i < (size_t)Nn * Dd) agg[i] = 0.0f;
  if (i < 256) stats[i] = 0.0f;  // column sum + column sumsq
}

// ---------------------------------------------------------- edge block ----
__global__ __launch_bounds__(256) void edge_block_kernel(
    const float* __restrict__ x, const float* __restrict__ eattr,
    const int* __restrict__ send, const int* __restrict__ recv,
    const __bf16* __restrict__ w1g, const float* __restrict__ b1,
    const __bf16* __restrict__ w2g, const float* __restrict__ b2,
    const __bf16* __restrict__ w3g, const float* __restrict__ b3,
    const float* __restrict__ gammag, const float* __restrict__ betag,
    float* __restrict__ eout, float* __restrict__ agg) {
  extern __shared__ unsigned char smem[];
  __bf16* W1l = (__bf16*)(smem + OFF_W1);
  __bf16* W2l = (__bf16*)(smem + OFF_W2);
  __bf16* W3l = (__bf16*)(smem + OFF_W3);
  __bf16* Al  = (__bf16*)(smem + OFF_A);
  __bf16* Hl  = (__bf16*)(smem + OFF_H);
  float* b1l  = (float*)(smem + OFF_B1);
  float* b2l  = (float*)(smem + OFF_B2);
  float* b3l  = (float*)(smem + OFF_B3);
  float* gam  = (float*)(smem + OFF_GAM);
  float* bet  = (float*)(smem + OFF_BET);
  int*   sidx = (int*)(smem + OFF_SIDX);
  int*   ridx = (int*)(smem + OFF_RIDX);
  float* psum = (float*)(smem + OFF_PS);
  float* psq  = (float*)(smem + OFF_PQ);

  const int tid = threadIdx.x, wave = tid >> 5, lane = tid & 31;
  const int e0 = blockIdx.x * 128;
  const int rbase = (wave & 3) * 32;       // local row block (2 row tiles)
  const int ch    = wave >> 2;             // column half (0:cols 0-63, 1:64-127)
  const int colb  = ch * 64;

  if (tid < 128) {
    sidx[tid] = send[e0 + tid];
    ridx[tid] = recv[e0 + tid];
    b1l[tid] = b1[tid];  b2l[tid] = b2[tid];  b3l[tid] = b3[tid];
    gam[tid] = gammag[tid];  bet[tid] = betag[tid];
  }
  __syncthreads();

  // Stage gathered edge input [128 edges][384] as bf16 (x[s] | x[r] | e)
  for (int i = tid; i < 128 * 384; i += 256) {
    const int er = i / 384, k = i - er * 384;
    float v;
    if (k < 128)      v = x[(size_t)sidx[er] * Dd + k];
    else if (k < 256) v = x[(size_t)ridx[er] * Dd + (k - 128)];
    else              v = eattr[(size_t)(e0 + er) * Dd + (k - 256)];
    Al[i] = (__bf16)v;
  }
  // Stage ALL layer weights once (resident for the whole block)
  for (int i = tid; i < 384 * 128; i += 256) W1l[i] = w1g[i];
  for (int i = tid; i < 128 * 128; i += 256) W2l[i] = w2g[i];
  for (int i = tid; i < 128 * 128; i += 256) W3l[i] = w3g[i];
  __syncthreads();

  v8f acc[2][4];
  zero_acc2(acc);
  gemm_rows32(Al + (size_t)rbase * 384, 384, W1l, colb, 12, lane, acc);  // L1 K=384
  silu_store2(acc, b1l, Hl, 128, rbase, colb, lane);
  __syncthreads();

  zero_acc2(acc);
  gemm_rows32(Hl + (size_t)rbase * 128, 128, W2l, colb, 4, lane, acc);   // L2 K=128
  silu_store2(acc, b2l, Al, 128, rbase, colb, lane);  // reuse Al as [128][128]
  __syncthreads();

  zero_acc2(acc);
  gemm_rows32(Al + (size_t)rbase * 128, 128, W3l, colb, 4, lane, acc);   // L3 K=128

  // ---- epilogue: bias + LayerNorm (cross-wave over column halves) ------
  const int l = lane & 15, mbase = (lane >> 4) * 8;
#pragma unroll
  for (int rt = 0; rt < 2; ++rt)
#pragma unroll
    for (int ct = 0; ct < 4; ++ct) {
      const float bia = b3l[colb + ct * 16 + l];
#pragma unroll
      for (int g = 0; g < 8; ++g) acc[rt][ct][g] += bia;
    }
  // per-row partials over this wave's 64 columns
#pragma unroll
  for (int rt = 0; rt < 2; ++rt)
#pragma unroll
    for (int g = 0; g < 8; ++g) {
      float s = 0.0f, q = 0.0f;
#pragma unroll
      for (int ct = 0; ct < 4; ++ct) { float v = acc[rt][ct][g]; s += v; q += v * v; }
#pragma unroll
      for (int off = 1; off < 16; off <<= 1) {
        s += __shfl_xor(s, off, 16);
        q += __shfl_xor(q, off, 16);
      }
      if (l == 0) {
        const int row = rbase + rt * 16 + mbase + g;
        psum[row * 2 + ch] = s;
        psq[row * 2 + ch] = q;
      }
    }
  __syncthreads();

  float mu[2][8], rstd[2][8];
#pragma unroll
  for (int rt = 0; rt < 2; ++rt)
#pragma unroll
    for (int g = 0; g < 8; ++g) {
      const int row = rbase + rt * 16 + mbase + g;
      const float S = psum[row * 2] + psum[row * 2 + 1];
      const float Q = psq[row * 2] + psq[row * 2 + 1];
      const float m_ = S * (1.0f / Dd);
      mu[rt][g] = m_;
      rstd[rt][g] = rsqrtf(Q * (1.0f / Dd) - m_ * m_ + 1e-5f);
    }

  // residual write + scatter-add of messages
#pragma unroll
  for (int rt = 0; rt < 2; ++rt)
#pragma unroll
    for (int ct = 0; ct < 4; ++ct) {
      const int col = colb + ct * 16 + l;
      const float gc = gam[col], bc = bet[col];
#pragma unroll
      for (int g = 0; g < 8; ++g) {
        const int row = rbase + rt * 16 + mbase + g;
        const float ln = (acc[rt][ct][g] - mu[rt][g]) * rstd[rt][g] * gc + bc;
        const size_t idx = (size_t)(e0 + row) * Dd + col;
        eout[idx] = eattr[idx] + ln;                                // residual
        atomicAdd(&agg[(size_t)ridx[row] * Dd + col], ln);          // segment_sum
      }
    }
}

// ---------------------------------------------------------- node block ----
__global__ __launch_bounds__(256) void node_block_kernel(
    const float* __restrict__ x, const float* __restrict__ agg,
    const __bf16* __restrict__ w1g, const float* __restrict__ b1,
    const __bf16* __restrict__ w2g, const float* __restrict__ b2,
    const __bf16* __restrict__ w3g, const float* __restrict__ b3,
    const float* __restrict__ gammag, const float* __restrict__ betag,
    float* __restrict__ xnew, float* __restrict__ gsum, float* __restrict__ gsq) {
  extern __shared__ unsigned char smem[];
  __bf16* W1l = (__bf16*)(smem + OFF_W1);   // 256x128 fits in W1 region
  __bf16* W2l = (__bf16*)(smem + OFF_W2);
  __bf16* W3l = (__bf16*)(smem + OFF_W3);
  __bf16* Al  = (__bf16*)(smem + OFF_A);
  __bf16* Hl  = (__bf16*)(smem + OFF_H);
  float* b1l  = (float*)(smem + OFF_B1);
  float* b2l  = (float*)(smem + OFF_B2);
  float* b3l  = (float*)(smem + OFF_B3);
  float* gam  = (float*)(smem + OFF_GAM);
  float* bet  = (float*)(smem + OFF_BET);
  float* csum = (float*)(smem + OFF_CSUM);
  float* csq  = (float*)(smem + OFF_CSQ);
  float* psum = (float*)(smem + OFF_PS);
  float* psq  = (float*)(smem + OFF_PQ);

  const int tid = threadIdx.x, wave = tid >> 5, lane = tid & 31;
  const int n0 = blockIdx.x * 128;
  const int rbase = (wave & 3) * 32;
  const int ch    = wave >> 2;
  const int colb  = ch * 64;

  if (tid < 128) {
    b1l[tid] = b1[tid];  b2l[tid] = b2[tid];  b3l[tid] = b3[tid];
    gam[tid] = gammag[tid];  bet[tid] = betag[tid];
    csum[tid] = 0.0f;  csq[tid] = 0.0f;
  }
  __syncthreads();

  // Stage [128 nodes][256] = (x | agg) as bf16 (clamp rows past N)
  for (int i = tid; i < 128 * 256; i += 256) {
    const int nr = i / 256, k = i - nr * 256;
    int node = n0 + nr; if (node >= Nn) node = Nn - 1;
    const float v = (k < 128) ? x[(size_t)node * Dd + k]
                              : agg[(size_t)node * Dd + (k - 128)];
    Al[i] = (__bf16)v;
  }
  for (int i = tid; i < 256 * 128; i += 256) W1l[i] = w1g[i];
  for (int i = tid; i < 128 * 128; i += 256) W2l[i] = w2g[i];
  for (int i = tid; i < 128 * 128; i += 256) W3l[i] = w3g[i];
  __syncthreads();

  v8f acc[2][4];
  zero_acc2(acc);
  gemm_rows32(Al + (size_t)rbase * 256, 256, W1l, colb, 8, lane, acc);   // L1 K=256
  silu_store2(acc, b1l, Hl, 128, rbase, colb, lane);
  __syncthreads();

  zero_acc2(acc);
  gemm_rows32(Hl + (size_t)rbase * 128, 128, W2l, colb, 4, lane, acc);   // L2
  silu_store2(acc, b2l, Al, 128, rbase, colb, lane);
  __syncthreads();

  zero_acc2(acc);
  gemm_rows32(Al + (size_t)rbase * 128, 128, W3l, colb, 4, lane, acc);   // L3

  // ---- epilogue: bias + LayerNorm + residual + PairNorm partials -------
  const int l = lane & 15, mbase = (lane >> 4) * 8;
#pragma unroll
  for (int rt = 0; rt < 2; ++rt)
#pragma unroll
    for (int ct = 0; ct < 4; ++ct) {
      const float bia = b3l[colb + ct * 16 + l];
#pragma unroll
      for (int g = 0; g < 8; ++g) acc[rt][ct][g] += bia;
    }
#pragma unroll
  for (int rt = 0; rt < 2; ++rt)
#pragma unroll
    for (int g = 0; g < 8; ++g) {
      float s = 0.0f, q = 0.0f;
#pragma unroll
      for (int ct = 0; ct < 4; ++ct) { float v = acc[rt][ct][g]; s += v; q += v * v; }
#pragma unroll
      for (int off = 1; off < 16; off <<= 1) {
        s += __shfl_xor(s, off, 16);
        q += __shfl_xor(q, off, 16);
      }
      if (l == 0) {
        const int row = rbase + rt * 16 + mbase + g;
        psum[row * 2 + ch] = s;
        psq[row * 2 + ch] = q;
      }
    }
  __syncthreads();

  float mu[2][8], rstd[2][8];
#pragma unroll
  for (int rt = 0; rt < 2; ++rt)
#pragma unroll
    for (int g = 0; g < 8; ++g) {
      const int row = rbase + rt * 16 + mbase + g;
      const float S = psum[row * 2] + psum[row * 2 + 1];
      const float Q = psq[row * 2] + psq[row * 2 + 1];
      const float m_ = S * (1.0f / Dd);
      mu[rt][g] = m_;
      rstd[rt][g] = rsqrtf(Q * (1.0f / Dd) - m_ * m_ + 1e-5f);
    }

#pragma unroll
  for (int rt = 0; rt < 2; ++rt)
#pragma unroll
    for (int ct = 0; ct < 4; ++ct) {
      const int col = colb + ct * 16 + l;
      const float gc = gam[col], bc = bet[col];
      float s = 0.0f, q = 0.0f;
#pragma unroll
      for (int g = 0; g < 8; ++g) {
        const int node = n0 + rbase + rt * 16 + mbase + g;
        const float ln = (acc[rt][ct][g] - mu[rt][g]) * rstd[rt][g] * gc + bc;
        if (node < Nn) {
          const size_t idx = (size_t)node * Dd + col;
          const float nv = x[idx] + ln;               // residual
          xnew[idx] = nv;
          s += nv; q += nv * nv;
        }
      }
      atomicAdd(&csum[col], s);
      atomicAdd(&csq[col], q);
    }
  __syncthreads();
  if (tid < 128) {
    atomicAdd(&gsum[tid], csum[tid]);
    atomicAdd(&gsq[tid], csq[tid]);
  }
}

// ------------------------------------------------------------ pairnorm ----
__global__ void pairnorm_stats(const float* __restrict__ gsum, const float* __restrict__ gsq,
                               float* __restrict__ mean, float* __restrict__ invr) {
  __shared__ float red[128];
  const int t = threadIdx.x;
  const float m = gsum[t] * (1.0f / (float)Nn);
  mean[t] = m;
  red[t] = gsq[t] - (float)Nn * m * m;   // sum of centered squares, per column
  __syncthreads();
  for (int off = 64; off > 0; off >>= 1) {
    if (t < off) red[t] += red[t + off];
    __syncthreads();
  }
  if (t == 0) invr[0] = 1.0f / (sqrtf(red[0] * (1.0f / (float)Nn)) + 1e-8f);
}

__global__ void pairnorm_apply(const float* __restrict__ xnew, const float* __restrict__ mean,
                               const float* __restrict__ invr, float* __restrict__ outx) {
  size_t i = (size_t)blockIdx.x * blockDim.x + threadIdx.x;
  if (i >= (size_t)Nn * Dd) return;
  outx[i] = (xnew[i] - mean[i & 127]) * invr[0];
}

// -------------------------------------------------------------- launch ----
extern "C" void kernel_launch(void* const* d_in, const int* in_sizes, int n_in,
                              void* d_out, int out_size, void* d_ws, size_t ws_size,
                              hipStream_t stream) {
  (void)in_sizes; (void)n_in; (void)out_size; (void)ws_size;

  const float* x     = (const float*)d_in[0];
  const float* eattr = (const float*)d_in[1];
  const int*   eidx  = (const int*)d_in[2];
  const float* ew1 = (const float*)d_in[3];  const float* eb1 = (const float*)d_in[4];
  const float* ew2 = (const float*)d_in[5];  const float* eb2 = (const float*)d_in[6];
  const float* ew3 = (const float*)d_in[7];  const float* eb3 = (const float*)d_in[8];
  const float* egam = (const float*)d_in[9]; const float* ebet = (const float*)d_in[10];
  const float* nw1 = (const float*)d_in[11]; const float* nb1 = (const float*)d_in[12];
  const float* nw2 = (const float*)d_in[13]; const float* nb2 = (const float*)d_in[14];
  const float* nw3 = (const float*)d_in[15]; const float* nb3 = (const float*)d_in[16];
  const float* ngam = (const float*)d_in[17]; const float* nbet = (const float*)d_in[18];

  float* out_x = (float*)d_out;                       // x_new      [N,128]
  float* out_e = out_x + (size_t)Nn * Dd;             // edge_attr' [E,128]

  const size_t ND = (size_t)Nn * Dd;
  float* agg  = (float*)d_ws;        // [N,128] message aggregation
  float* xnew = agg + ND;            // [N,128] pre-PairNorm nodes
  float* gsum = xnew + ND;           // [128] column sums
  float* gsq  = gsum + Dd;           // [128] column sum-squares
  float* mean = gsq + Dd;            // [128]
  float* invr = mean + Dd;           // [1]
  __bf16* wbf = (__bf16*)(((uintptr_t)(invr + 1) + 15) & ~(uintptr_t)15);
  __bf16* we1 = wbf;                 // 384*128
  __bf16* we2 = we1 + 384 * 128;     // 128*128
  __bf16* we3 = we2 + 128 * 128;     // 128*128
  __bf16* wn1 = we3 + 128 * 128;     // 256*128
  __bf16* wn2 = wn1 + 256 * 128;     // 128*128
  __bf16* wn3 = wn2 + 128 * 128;     // 128*128

  hipFuncSetAttribute(reinterpret_cast<const void*>(edge_block_kernel),
                      hipFuncAttributeMaxDynamicSharedMemorySize, SMEM_BYTES);
  hipFuncSetAttribute(reinterpret_cast<const void*>(node_block_kernel),
                      hipFuncAttributeMaxDynamicSharedMemorySize, SMEM_BYTES);

  const int totW = (384 + 128 + 128 + 256 + 128 + 128) * 128;  // 147456
  prep_weights<<<(totW + 255) / 256, 256, 0, stream>>>(ew1, ew2, ew3, nw1, nw2, nw3, wbf);
  zero_ws<<<(int)((ND + 255) / 256), 256, 0, stream>>>(agg, gsum);

  edge_block_kernel<<<Ee / 128, 256, SMEM_BYTES, stream>>>(
      x, eattr, eidx, eidx + Ee,
      we1, eb1, we2, eb2, we3, eb3, egam, ebet, out_e, agg);

  node_block_kernel<<<(Nn + 127) / 128, 256, SMEM_BYTES, stream>>>(
      x, agg, wn1, nb1, wn2, nb2, wn3, nb3, ngam, nbet, xnew, gsum, gsq);

  pairnorm_stats<<<1, 128, 0, stream>>>(gsum, gsq, mean, invr);
  pairnorm_apply<<<(int)((ND + 255) / 256), 256, 0, stream>>>(xnew, mean, invr, out_x);
}
